// AttentionBlock_2362232013224
// MI455X (gfx1250) — matrix operand order repaired and checked by reading, not compiled, not run on hardware
//
#include <hip/hip_runtime.h>
#include <hip/hip_bf16.h>

typedef __bf16 bf16_t;
typedef __bf16 v16bf __attribute__((ext_vector_type(16)));
typedef __bf16 v8bf  __attribute__((ext_vector_type(8)));
typedef float  v8f   __attribute__((ext_vector_type(8)));

// Payload type for the async global->LDS b128 builtin, exactly as clang
// declares it: 'int __attribute__((vector_size(16)))' with as1/as3 pointers.
typedef int v4i_pay __attribute__((vector_size(16)));
typedef __attribute__((address_space(1))) v4i_pay as1_v4i;
typedef __attribute__((address_space(3))) v4i_pay as3_v4i;

#if defined(__HIP_DEVICE_COMPILE__) && __has_builtin(__builtin_amdgcn_global_load_async_to_lds_b128)
#define USE_ASYNC_LDS 1
#else
#define USE_ASYNC_LDS 0
#endif

#if defined(__HIP_DEVICE_COMPILE__) && __has_builtin(__builtin_amdgcn_s_wait_asynccnt)
#define WAIT_ASYNC() __builtin_amdgcn_s_wait_asynccnt(0)
#elif defined(__HIP_DEVICE_COMPILE__)
#define WAIT_ASYNC() asm volatile("s_wait_asynccnt 0x0" ::: "memory")
#else
#define WAIT_ASYNC()
#endif

#if USE_ASYNC_LDS
#define COPY_B128(ldsp, gp)                                                   \
  __builtin_amdgcn_global_load_async_to_lds_b128((as1_v4i*)(gp),              \
                                                 (as3_v4i*)(ldsp), 0, 0)
#else
#define COPY_B128(ldsp, gp) *(v8bf*)(ldsp) = *(const v8bf*)(gp)
#endif

// ---------------------------------------------------------------------------
// f32 -> bf16 conversion (for weights)
// ---------------------------------------------------------------------------
__global__ void __launch_bounds__(256) f32_to_bf16_kernel(
    const float* __restrict__ in, bf16_t* __restrict__ out, int n) {
  int i = blockIdx.x * 256 + threadIdx.x;
  if (i < n) out[i] = (bf16_t)in[i];
}

// ---------------------------------------------------------------------------
// GroupNorm over (B=16, C=512, N=4096), 8 groups.
// Writes TRANSPOSED bf16 activations: h_t[b][n][c]  (n-major, c contiguous)
// so downstream GEMM staging is fully K-contiguous.
// One block per (b, g).
// ---------------------------------------------------------------------------
__global__ void __launch_bounds__(256) group_norm_t_bf16_kernel(
    const float* __restrict__ x, const float* __restrict__ gamma,
    const float* __restrict__ beta, bf16_t* __restrict__ h_t) {
  __shared__ float redS[256];
  __shared__ float redQ[256];

  const int bg  = blockIdx.x;
  const int b   = bg >> 3;
  const int g   = bg & 7;
  const int tid = threadIdx.x;

  const float* xp = x + ((size_t)b * 512 + (size_t)g * 64) * 4096;

  float s = 0.f, q = 0.f;
  for (int ch = 0; ch < 64; ++ch) {
    const float* row = xp + (size_t)ch * 4096;
    for (int j = tid * 4; j < 4096; j += 1024) {
      float4 v = *(const float4*)(row + j);
      s += v.x + v.y + v.z + v.w;
      q += v.x * v.x + v.y * v.y + v.z * v.z + v.w * v.w;
    }
  }
  redS[tid] = s;
  redQ[tid] = q;
  __syncthreads();
  for (int off = 128; off > 0; off >>= 1) {
    if (tid < off) {
      redS[tid] += redS[tid + off];
      redQ[tid] += redQ[tid + off];
    }
    __syncthreads();
  }
  const float inv_n = 1.f / 262144.f;
  const float mean  = redS[0] * inv_n;
  const float var   = redQ[0] * inv_n - mean * mean;
  const float rstd  = rsqrtf(var + 1e-5f);

  // transposed write: for each spatial n, 64 contiguous channels (128B)
  for (int n = tid; n < 4096; n += 256) {
    bf16_t* orow = h_t + ((size_t)b * 4096 + n) * 512 + g * 64;
#pragma unroll
    for (int cg = 0; cg < 64; cg += 8) {
      union { bf16_t e[8]; uint4 u; } p;
#pragma unroll
      for (int i = 0; i < 8; ++i) {
        const float gm = gamma[g * 64 + cg + i] * rstd;
        const float bt = beta[g * 64 + cg + i] - mean * gm;
        p.e[i] = (bf16_t)(xp[(size_t)(cg + i) * 4096 + n] * gm + bt);
      }
      *(uint4*)(orow + cg) = p.u;
    }
  }
}

// ---------------------------------------------------------------------------
// Generic bf16 WMMA GEMM with K-contiguous operands:
//   C[z](M x N) = A[z](M x K, row-major) * Bt[z](N x K, row-major)^T
// Wave grid WMxWN (8 waves), wave tile (MT*16)x(NT*16), K-step 32.
// Both tiles staged via async global->LDS b128 DMA (no ds_store at all),
// double-buffered. BIAS_MODE: 0 none, 1 by output row (m), 2 by output col (n).
// ---------------------------------------------------------------------------
template <int ZDIV, int WM_, int WN_, int MT, int NT, int BIAS_MODE,
          bool HAS_RESID, bool C_F32>
__global__ void __launch_bounds__(256) wmma_gemm_tt(
    const bf16_t* __restrict__ A, const bf16_t* __restrict__ Bt,
    void* __restrict__ C, const float* __restrict__ bias,
    const float* __restrict__ resid, int K, int lda, int ldb, int ldc,
    long A_so, long A_si, long B_so, long B_si, long C_so, long C_si,
    long R_so, long R_si) {
  constexpr int BM = WM_ * MT * 16;
  constexpr int BN = WN_ * NT * 16;
  constexpr int BK = 32;
  __shared__ bf16_t sA[2][BM * BK];   // [m][k], k contiguous
  __shared__ bf16_t sB[2][BN * BK];   // [n][k], k contiguous

  const int tid = threadIdx.x;
  const int z   = blockIdx.z;
  const int zo  = z / ZDIV;
  const int zi  = z % ZDIV;
  const bf16_t* Ab  = A  + (long)zo * A_so + (long)zi * A_si;
  const bf16_t* Btb = Bt + (long)zo * B_so + (long)zi * B_si;
  const long coff   = (long)zo * C_so + (long)zi * C_si;
  const long roff   = (long)zo * R_so + (long)zi * R_si;

  const int n0 = blockIdx.x * BN;
  const int m0 = blockIdx.y * BM;
  const bf16_t* gA = Ab  + (size_t)m0 * lda;
  const bf16_t* gB = Btb + (size_t)n0 * ldb;

  const int w    = tid >> 5;
  const int lane = tid & 31;
  const int half = lane >> 4;
  const int l15  = lane & 15;
  const int wm   = w / WN_;
  const int wn   = w % WN_;

  const v8f vzero = {0.f, 0.f, 0.f, 0.f, 0.f, 0.f, 0.f, 0.f};
  v8f acc[MT][NT];
#pragma unroll
  for (int mt = 0; mt < MT; ++mt)
#pragma unroll
    for (int nt = 0; nt < NT; ++nt) acc[mt][nt] = vzero;

  constexpr int A_CH = BM * 4;   // v8bf (16B) chunks per A tile
  constexpr int B_CH = BN * 4;

  auto stage = [&](int buf, int kc) {
#pragma unroll
    for (int c = tid; c < A_CH; c += 256) {
      const int r = c >> 2, ck = (c & 3) * 8;
      COPY_B128(sA[buf] + r * BK + ck, gA + (size_t)r * lda + kc + ck);
    }
#pragma unroll
    for (int c = tid; c < B_CH; c += 256) {
      const int r = c >> 2, ck = (c & 3) * 8;
      COPY_B128(sB[buf] + r * BK + ck, gB + (size_t)r * ldb + kc + ck);
    }
  };

  const int T = K / BK;

  stage(0, 0);
  WAIT_ASYNC();
  __syncthreads();

  for (int t = 0; t < T; ++t) {
    const int cur = t & 1;
    const int nxt = cur ^ 1;
    const bool more = (t + 1) < T;

    if (more) stage(nxt, (t + 1) * BK);   // async DMA into the free buffer

    v16bf af[MT], bfr[NT];
#pragma unroll
    for (int mt = 0; mt < MT; ++mt) {
      const bf16_t* ap = sA[cur] + (wm * MT * 16 + mt * 16 + l15) * BK;
      v8bf lo = *(const v8bf*)(ap + half * 8);        // K = half*8 .. +7
      v8bf hi = *(const v8bf*)(ap + 16 + half * 8);   // K = 16+half*8 .. +7
      af[mt] = __builtin_shufflevector(lo, hi, 0, 1, 2, 3, 4, 5, 6, 7, 8, 9,
                                       10, 11, 12, 13, 14, 15);
    }
#pragma unroll
    for (int nt = 0; nt < NT; ++nt) {
      const bf16_t* bp = sB[cur] + (wn * NT * 16 + nt * 16 + l15) * BK;
      v8bf blo = *(const v8bf*)(bp + half * 8);
      v8bf bhi = *(const v8bf*)(bp + 16 + half * 8);
      bfr[nt] = __builtin_shufflevector(blo, bhi, 0, 1, 2, 3, 4, 5, 6, 7, 8, 9,
                                        10, 11, 12, 13, 14, 15);
    }
#pragma unroll
    for (int mt = 0; mt < MT; ++mt)
#pragma unroll
      for (int nt = 0; nt < NT; ++nt)
        acc[mt][nt] = __builtin_amdgcn_wmma_f32_16x16x32_bf16(
            false, af[mt], false, bfr[nt], (short)0, acc[mt][nt], false, false);

    if (more) WAIT_ASYNC();
    __syncthreads();
  }

  // branch-free epilogue
#pragma unroll
  for (int mt = 0; mt < MT; ++mt)
#pragma unroll
    for (int nt = 0; nt < NT; ++nt) {
      const int gm0 = m0 + wm * MT * 16 + mt * 16 + half * 8;
      const int gn  = n0 + wn * NT * 16 + nt * 16 + l15;
      const long cb = coff + (long)gm0 * ldc + gn;
      const long rb = roff + (long)gm0 * ldc + gn;
#pragma unroll
      for (int r = 0; r < 8; ++r) {
        float v = acc[mt][nt][r];
        if (BIAS_MODE == 1) v += bias[gm0 + r];
        if (BIAS_MODE == 2) v += bias[gn];
        if (HAS_RESID)      v += resid[rb + (long)r * ldc];
        if (C_F32) ((float*)C)[cb + (long)r * ldc] = v;
        else       ((bf16_t*)C)[cb + (long)r * ldc] = (bf16_t)v;
      }
    }
}

// ---------------------------------------------------------------------------
// Fused attention scores + softmax per (b, h), reading qkv_t[n][o]:
//   att[d,e] = softmax_e( scale * sum_n q[d,n] * k[e,n] )
// q[d,n] = qkv_t[n][h*64+d], k[e,n] = qkv_t[n][512+h*64+e].
// Small transpose scatter into LDS (only ~6% of total FLOPs live here).
// ---------------------------------------------------------------------------
__global__ void __launch_bounds__(256) attn_scores_softmax_kernel(
    const bf16_t* __restrict__ qkv_t, bf16_t* __restrict__ att, float scale) {
  __shared__ bf16_t sQ[64 * 32];   // [d][n_local]
  __shared__ bf16_t sK[64 * 32];   // [e][n_local]
  __shared__ float  sAtt[64 * 64];

  const int z = blockIdx.x;     // b*8 + h
  const int b = z >> 3;
  const int h = z & 7;
  const bf16_t* qt = qkv_t + (size_t)b * 4096 * 1536;
  const int qcol = h * 64;
  const int kcol = 512 + h * 64;

  const int tid  = threadIdx.x;
  const int w    = tid >> 5;
  const int lane = tid & 31;
  const int half = lane >> 4;
  const int l15  = lane & 15;
  const int wm   = w >> 2;
  const int wn   = w & 3;

  const v8f vzero = {0.f, 0.f, 0.f, 0.f, 0.f, 0.f, 0.f, 0.f};
  v8f acc[2];
  acc[0] = vzero; acc[1] = vzero;

  // staging: thread -> (n_local 0..31, 8-wide d/e chunk)
  const int nl = tid >> 3;
  const int dc = (tid & 7) * 8;

  for (int kc = 0; kc < 4096; kc += 32) {
    const bf16_t* rowp = qt + (size_t)(kc + nl) * 1536;
    v8bf vq = *(const v8bf*)(rowp + qcol + dc);
    v8bf vk = *(const v8bf*)(rowp + kcol + dc);
#pragma unroll
    for (int i = 0; i < 8; ++i) {
      sQ[(dc + i) * 32 + nl] = vq[i];
      sK[(dc + i) * 32 + nl] = vk[i];
    }
    __syncthreads();

    const bf16_t* bp = sK + (wn * 16 + l15) * 32;
    v8bf blo = *(const v8bf*)(bp + half * 8);
    v8bf bhi = *(const v8bf*)(bp + 16 + half * 8);
    v16bf bfrag = __builtin_shufflevector(blo, bhi, 0, 1, 2, 3, 4, 5, 6, 7, 8,
                                          9, 10, 11, 12, 13, 14, 15);
#pragma unroll
    for (int mt = 0; mt < 2; ++mt) {
      const bf16_t* ap = sQ + (wm * 32 + mt * 16 + l15) * 32;
      v8bf lo = *(const v8bf*)(ap + half * 8);
      v8bf hi = *(const v8bf*)(ap + 16 + half * 8);
      v16bf afrag = __builtin_shufflevector(lo, hi, 0, 1, 2, 3, 4, 5, 6, 7, 8,
                                            9, 10, 11, 12, 13, 14, 15);
      acc[mt] = __builtin_amdgcn_wmma_f32_16x16x32_bf16(
          false, afrag, false, bfrag, (short)0, acc[mt], false, false);
    }
    __syncthreads();
  }

#pragma unroll
  for (int mt = 0; mt < 2; ++mt)
#pragma unroll
    for (int r = 0; r < 8; ++r) {
      const int d = wm * 32 + mt * 16 + r + half * 8;
      const int e = wn * 16 + l15;
      sAtt[d * 64 + e] = acc[mt][r] * scale;
    }
  __syncthreads();

  if (tid < 64) {
    const float* rp = sAtt + tid * 64;
    float m = -3.4e38f;
    for (int e = 0; e < 64; ++e) m = fmaxf(m, rp[e]);
    float s = 0.f;
    for (int e = 0; e < 64; ++e) s += __expf(rp[e] - m);
    const float inv = 1.f / s;
    bf16_t* arow = att + (size_t)z * 4096 + (size_t)tid * 64;
    for (int e = 0; e < 64; ++e) arow[e] = (bf16_t)(__expf(rp[e] - m) * inv);
  }
}

// ---------------------------------------------------------------------------
// Host-side orchestration
// ---------------------------------------------------------------------------
extern "C" void kernel_launch(void* const* d_in, const int* in_sizes, int n_in,
                              void* d_out, int out_size, void* d_ws,
                              size_t ws_size, hipStream_t stream) {
  (void)in_sizes; (void)n_in; (void)out_size; (void)ws_size;
  const float* x     = (const float*)d_in[0];
  const float* gamma = (const float*)d_in[1];
  const float* beta  = (const float*)d_in[2];
  const float* w_qkv = (const float*)d_in[3];
  const float* b_qkv = (const float*)d_in[4];
  const float* w_out = (const float*)d_in[5];
  const float* b_out = (const float*)d_in[6];
  float* out = (float*)d_out;

  char* ws = (char*)d_ws;
  size_t off = 0;
  auto take = [&](size_t bytes) -> char* {
    char* p = ws + off;
    off += (bytes + 255) & ~(size_t)255;
    return p;
  };
  bf16_t* h_t     = (bf16_t*)take((size_t)16 * 4096 * 512 * 2);   // reused as hv_t
  bf16_t* qkv_t   = (bf16_t*)take((size_t)16 * 4096 * 1536 * 2);
  bf16_t* att_bf  = (bf16_t*)take((size_t)16 * 8 * 64 * 64 * 2);
  bf16_t* wqkv_bf = (bf16_t*)take((size_t)1536 * 512 * 2);
  bf16_t* wout_bf = (bf16_t*)take((size_t)512 * 512 * 2);

  f32_to_bf16_kernel<<<(1536 * 512 + 255) / 256, 256, 0, stream>>>(
      w_qkv, wqkv_bf, 1536 * 512);
  f32_to_bf16_kernel<<<(512 * 512 + 255) / 256, 256, 0, stream>>>(
      w_out, wout_bf, 512 * 512);

  // GroupNorm -> transposed bf16 h_t[b][n][c]
  group_norm_t_bf16_kernel<<<128, 256, 0, stream>>>(x, gamma, beta, h_t);

  // QKV (transposed product): qkv_t[b][n][o] = h_t(n x c) . w_qkv^T + b_qkv[o]
  // M = 4096 (n), N = 1536 (o), K = 512 (c); Bt = w_qkv rows (o x c).
  wmma_gemm_tt<1, 2, 4, 2, 2, /*bias_n*/2, false, false>
      <<<dim3(1536 / 128, 4096 / 64, 16), 256, 0, stream>>>(
          h_t, wqkv_bf, qkv_t, b_qkv, nullptr,
          512, 512, 512, 1536,
          (long)4096 * 512, 0L, 0L, 0L, (long)4096 * 1536, 0L, 0L, 0L);

  // scores + softmax per (b, h): scale = 1/sqrt(64) = 0.125
  attn_scores_softmax_kernel<<<128, 256, 0, stream>>>(qkv_t, att_bf, 0.125f);

  // hv (transposed product) per (b, h):
  //   hv_t[n][h*64+d] = sum_e v_t[n][e] * att[d][e]
  // M = 4096 (n), N = 64 (d), K = 64 (e); A = qkv_t cols [1024+h*64 ..),
  // Bt = att rows (d x e). BM=64, BN=64 (WM=2,WN=4,MT=2,NT=1).
  wmma_gemm_tt<8, 2, 4, 2, 1, 0, false, false>
      <<<dim3(1, 4096 / 64, 128), 256, 0, stream>>>(
          qkv_t + (size_t)1024, att_bf, h_t, nullptr, nullptr,
          64, 1536, 64, 512,
          (long)4096 * 1536, 64L,          // A: batch stride, head col offset
          (long)8 * 64 * 64, (long)64 * 64, // Bt: att per batch / per head
          (long)4096 * 512, 64L,            // C: hv_t batch stride, head col
          0L, 0L);

  // out projection (normal orientation) + bias + residual -> f32 d_out:
  //   out[o][n] = sum_c w_out[o][c] * hv_t[n][c] + b_out[o] + x[o][n]
  // M = 512 (o), N = 4096 (n), K = 512 (c); Bt = hv_t rows (n x c).
  wmma_gemm_tt<1, 2, 4, 2, 2, /*bias_m*/1, true, true>
      <<<dim3(4096 / 128, 512 / 64, 16), 256, 0, stream>>>(
          wout_bf, h_t, out, b_out, x,
          512, 512, 512, 4096,
          0L, 0L, (long)4096 * 512, 0L, (long)512 * 4096, 0L,
          (long)512 * 4096, 0L);
}
